// Int4MarlinLinear_54975581389478
// MI455X (gfx1250) — compile-verified, matrix operand order checked by reading, not run on hardware
//
#include <hip/hip_runtime.h>

// Problem constants (match reference)
#define NI4   6144
#define NI8   2048
#define NTOT  8192      // NI4 + NI8
#define KK    8192
#define GRP   128
#define MM    16

typedef __attribute__((ext_vector_type(16))) _Float16 v16h;
typedef __attribute__((ext_vector_type(8)))  _Float16 h8;
typedef __attribute__((ext_vector_type(8)))  float    v8f;
typedef __attribute__((ext_vector_type(4)))  int      i4v;

// ---------------------------------------------------------------------------
// Kernel 1: x2f[m,k] = (f16)( x[m,k] / awq_scales[k] )   (16 x 8192 panel)
// ---------------------------------------------------------------------------
__global__ void prep_x2_kernel(const float* __restrict__ x,
                               const float* __restrict__ awq,
                               _Float16* __restrict__ x2f) {
    int idx = blockIdx.x * blockDim.x + threadIdx.x;   // 0 .. 16*8192-1
    int k   = idx & (KK - 1);
    x2f[idx] = (_Float16)(x[idx] / awq[k]);
}

// ---------------------------------------------------------------------------
// Kernel 2: WMMA GEMM over dequantized weights.
// One block = one 16-wide column tile; 4 waves split K 4-ways; LDS-reduce.
//
// Fragment layout (wave32, v_wmma_f32_16x16x32_f16):
//   A (16x32 f16): lane l supplies row m = l&15; K within chunk:
//     VGPR 0-3 hold K = (l>>4)*8 + 0..7, VGPR 4-7 hold K = 16 + (l>>4)*8 + 0..7
//   B (32x16 f16): symmetric — lane l supplies column n = l&15 with the
//     same K striping, i.e. one weight row striped across the lane's VGPRs.
//   C/D (16x16 f32): VGPR r, lane l -> element (m = r + 8*(l>>4), n = l&15)
// ---------------------------------------------------------------------------
__global__ __launch_bounds__(128)
void qgemm_wmma_kernel(const _Float16* __restrict__ x2f,
                       const int*      __restrict__ w4,
                       const float*    __restrict__ s4,
                       const int*      __restrict__ w8,
                       const float*    __restrict__ s8,
                       float*          __restrict__ ycat) {
    __shared__ float red[4 * 256];

    const int tile  = blockIdx.x;            // 0..511
    const int lane  = threadIdx.x & 31;
    const int wave  = threadIdx.x >> 5;      // 0..3 (K split)
    const int halfL = lane >> 4;             // K-half selector
    const int nloc  = lane & 15;             // A row AND B column index
    const int col   = tile * 16 + nloc;      // output column (wave-uniform tile)

    const int kbase = wave * (KK / 4);
    const _Float16* arow = x2f + (size_t)nloc * KK;

    v8f c = {};

    if (tile * 16 < NI4) {
        // ---------------- int4 region: B = (w - 8) * s4[group] ----------------
        const int*   wrow = w4 + (size_t)col * KK;
        const float* srow = s4 + (size_t)col * (KK / GRP);
        for (int k0 = kbase; k0 < kbase + KK / 4; k0 += 32) {
            __builtin_prefetch(wrow + k0 + 512, 0, 1);    // global_prefetch next lines
            // A fragment: two 16-byte loads of the activation panel (L2 hit)
            h8 alo = *(const h8*)(arow + k0 + halfL * 8);
            h8 ahi = *(const h8*)(arow + k0 + 16 + halfL * 8);
            v16h a;
            #pragma unroll
            for (int i = 0; i < 8; ++i) { a[i] = alo[i]; a[8 + i] = ahi[i]; }
            // B fragment: 16 int32 weights via 4x global_load_b128, dequant f16
            const float s = srow[k0 >> 7];                // one group per 32-chunk
            i4v w0 = *(const i4v*)(wrow + k0 + halfL * 8);
            i4v w1 = *(const i4v*)(wrow + k0 + halfL * 8 + 4);
            i4v w2 = *(const i4v*)(wrow + k0 + 16 + halfL * 8);
            i4v w3 = *(const i4v*)(wrow + k0 + 16 + halfL * 8 + 4);
            v16h b;
            #pragma unroll
            for (int i = 0; i < 4; ++i) {
                b[i]      = (_Float16)((float)(w0[i] - 8) * s);
                b[4 + i]  = (_Float16)((float)(w1[i] - 8) * s);
                b[8 + i]  = (_Float16)((float)(w2[i] - 8) * s);
                b[12 + i] = (_Float16)((float)(w3[i] - 8) * s);
            }
            c = __builtin_amdgcn_wmma_f32_16x16x32_f16(false, a, false, b,
                                                       (short)0, c, false, false);
        }
    } else {
        // ------- int8 region: B = (w - 128); row scale s8 applied after -------
        // (the -128 offset exactly absorbs the 128*s8*sum_x correction term)
        const int* wrow = w8 + (size_t)(col - NI4) * KK;
        for (int k0 = kbase; k0 < kbase + KK / 4; k0 += 32) {
            __builtin_prefetch(wrow + k0 + 512, 0, 1);
            h8 alo = *(const h8*)(arow + k0 + halfL * 8);
            h8 ahi = *(const h8*)(arow + k0 + 16 + halfL * 8);
            v16h a;
            #pragma unroll
            for (int i = 0; i < 8; ++i) { a[i] = alo[i]; a[8 + i] = ahi[i]; }
            i4v w0 = *(const i4v*)(wrow + k0 + halfL * 8);
            i4v w1 = *(const i4v*)(wrow + k0 + halfL * 8 + 4);
            i4v w2 = *(const i4v*)(wrow + k0 + 16 + halfL * 8);
            i4v w3 = *(const i4v*)(wrow + k0 + 16 + halfL * 8 + 4);
            v16h b;
            #pragma unroll
            for (int i = 0; i < 4; ++i) {
                b[i]      = (_Float16)(w0[i] - 128);   // exact in f16
                b[4 + i]  = (_Float16)(w1[i] - 128);
                b[8 + i]  = (_Float16)(w2[i] - 128);
                b[12 + i] = (_Float16)(w3[i] - 128);
            }
            c = __builtin_amdgcn_wmma_f32_16x16x32_f16(false, a, false, b,
                                                       (short)0, c, false, false);
        }
    }

    // Spill per-wave partial 16x16 tile to LDS
    #pragma unroll
    for (int r = 0; r < 8; ++r) {
        int m = r + 8 * halfL;
        red[wave * 256 + m * 16 + nloc] = c[r];
    }
    __syncthreads();

    // 128 threads reduce 256 outputs across the 4 K-split waves
    for (int o = threadIdx.x; o < 256; o += 128) {
        float v = red[o] + red[256 + o] + red[512 + o] + red[768 + o];
        int m  = o >> 4;
        int n  = o & 15;
        int cc = tile * 16 + n;
        if (cc >= NI4) v *= s8[cc - NI4];     // deferred int8 row scale
        ycat[(size_t)m * NTOT + cc] = v;
    }
}

// ---------------------------------------------------------------------------
// Kernel 3: out[m,j] = ycat[m, inv_perm[j]] + bias[j]   (deterministic gather)
// ---------------------------------------------------------------------------
__global__ void gather_bias_kernel(const float* __restrict__ ycat,
                                   const int*   __restrict__ inv_perm,
                                   const float* __restrict__ bias,
                                   float*       __restrict__ out) {
    int idx = blockIdx.x * blockDim.x + threadIdx.x;   // 0 .. 16*8192-1
    int j = idx & (NTOT - 1);
    int m = idx >> 13;
    out[idx] = ycat[(size_t)m * NTOT + inv_perm[j]] + bias[j];
}

// ---------------------------------------------------------------------------
// Launch
// inputs: x, w_int4, s_int4, w_uint8, s_int8, awq_scales, bias, inv_perm, group_size
// ---------------------------------------------------------------------------
extern "C" void kernel_launch(void* const* d_in, const int* in_sizes, int n_in,
                              void* d_out, int out_size, void* d_ws, size_t ws_size,
                              hipStream_t stream) {
    const float* x    = (const float*)d_in[0];
    const int*   w4   = (const int*)  d_in[1];
    const float* s4   = (const float*)d_in[2];
    const int*   w8   = (const int*)  d_in[3];
    const float* s8   = (const float*)d_in[4];
    const float* awq  = (const float*)d_in[5];
    const float* bias = (const float*)d_in[6];
    const int*   iperm= (const int*)  d_in[7];
    float*       out  = (float*)d_out;

    // Workspace layout: [0, 256KB) f16 activation panel; [256KB, 768KB) f32 ycat
    _Float16* x2f  = (_Float16*)d_ws;
    float*    ycat = (float*)((char*)d_ws + (size_t)MM * KK * sizeof(_Float16));

    prep_x2_kernel<<<(MM * KK) / 256, 256, 0, stream>>>(x, awq, x2f);
    qgemm_wmma_kernel<<<NTOT / 16, 128, 0, stream>>>(x2f, w4, s4, w8, s8, ycat);
    gather_bias_kernel<<<(MM * NTOT) / 256, 256, 0, stream>>>(ycat, iperm, bias, out);
}
